// LabelSmoothLoss_58789512348383
// MI455X (gfx1250) — matrix operand order repaired and checked by reading, not compiled, not run on hardware
//
#include <hip/hip_runtime.h>
#include <hip/hip_bf16.h>
#include <math.h>

typedef __attribute__((ext_vector_type(16))) _Float16 v16h;
typedef __attribute__((ext_vector_type(8)))  _Float16 h8;
typedef __attribute__((ext_vector_type(2)))  _Float16 h2;
typedef __attribute__((ext_vector_type(8)))  float    v8f;

#define BDIM 256
#define LDIM 4096
#define A_STR 40   // 40 halves = 80 B = 5*16 B: b128-aligned rows, bank-optimal
#define BT_STR 40

// ---------------------------------------------------------------------------
// Kernel 1: rinv[lj] = 1 / (sum_li mask[li,lj] + (1 - mask[lj,lj]))
// 4 columns per thread via packed dword mask reads (bool = 1 byte, 0/1).
// ---------------------------------------------------------------------------
__global__ __launch_bounds__(256) void k_relnum(const unsigned char* __restrict__ adjm,
                                                float* __restrict__ rinv) {
    int lj0 = (blockIdx.x * 256 + threadIdx.x) * 4;   // grid 4 x 256 -> 4096 cols
    int c0 = 0, c1 = 0, c2 = 0, c3 = 0;
    for (int li = 0; li < LDIM; ++li) {
        unsigned v = __builtin_nontemporal_load(
            (const unsigned*)(adjm + (size_t)li * LDIM + lj0));
        c0 += (int)(v & 0xffu);
        c1 += (int)((v >> 8) & 0xffu);
        c2 += (int)((v >> 16) & 0xffu);
        c3 += (int)((v >> 24) & 0xffu);
    }
#pragma unroll
    for (int j = 0; j < 4; ++j) {
        int lj = lj0 + j;
        int s = (j == 0) ? c0 : (j == 1) ? c1 : (j == 2) ? c2 : c3;
        float diag = (float)adjm[(size_t)lj * LDIM + lj];
        rinv[lj] = 1.0f / ((float)s + 1.0f - diag);
    }
}

// ---------------------------------------------------------------------------
// Kernel 2: candidates = (predicts @ W) * rinv, W fused from adj_values/mask.
// Block: 256 threads (8 waves), computes all 256 rows x 64 cols.
// grid.x = L/64 = 64 -> adjacency streamed exactly once device-wide (NT).
// ---------------------------------------------------------------------------
__global__ __launch_bounds__(256) void k_cand(const float* __restrict__ predicts,
                                              const float* __restrict__ adjv,
                                              const unsigned char* __restrict__ adjm,
                                              const float* __restrict__ rinv,
                                              _Float16* __restrict__ candH) {
    __shared__ __align__(16) _Float16 aT[256 * A_STR];  // predicts tile 256x32
    __shared__ __align__(16) _Float16 bT[64 * BT_STR];  // W tile transposed [n][k]

    const int t     = threadIdx.x;
    const int lane  = t & 31;
    const int w     = t >> 5;        // wave 0..7
    const int n0    = blockIdx.x * 64;
    const int nb    = (w & 3) * 16;  // wave's column group in the 64-wide strip
    const int mrow  = lane & 15;
    const int khalf = lane >> 4;

    const int wn    = t & 63;        // W-tile: this thread's column
    const int wk    = (t >> 6) * 8;  // W-tile: this thread's K run (8 wide)

    const v8f vzero = {0.f, 0.f, 0.f, 0.f, 0.f, 0.f, 0.f, 0.f};
    v8f acc[8];
#pragma unroll
    for (int j = 0; j < 8; ++j) acc[j] = vzero;

    for (int k0 = 0; k0 < LDIM; k0 += 32) {
        // Stage predicts 256x32 -> f16 LDS: float2 global loads, packed h2 stores
#pragma unroll
        for (int i = 0; i < 16; ++i) {
            int e   = i * 256 + t;
            int row = e >> 4;
            int kd  = e & 15;
            float2 p = *(const float2*)(predicts + (size_t)row * LDIM + k0 + 2 * kd);
            h2 hp;
            hp[0] = (_Float16)p.x;
            hp[1] = (_Float16)p.y;
            *(h2*)(aT + row * A_STR + 2 * kd) = hp;
        }
        // Build W tile 32(K) x 64(N): NT global stream, one ds_store_b128/thread
        {
            h8 wv8;
#pragma unroll
            for (int i = 0; i < 8; ++i) {
                int k = wk + i;
                size_t gi = (size_t)(k0 + k) * LDIM + (size_t)(n0 + wn);
                unsigned char mm = __builtin_nontemporal_load(adjm + gi);
                float av = __builtin_nontemporal_load(adjv + gi);
                float wvv = mm ? av : (((k0 + k) == (n0 + wn)) ? 1.0f : 0.0f);
                wv8[i] = (_Float16)wvv;
            }
            *(h8*)(bT + wn * BT_STR + wk) = wv8;
            // Prefetch next K-step's adjacency tile
            if (k0 + 32 < LDIM) {
                size_t gp = (size_t)(k0 + 32 + wk) * LDIM + (size_t)(n0 + wn);
                __builtin_prefetch(adjv + gp, 0, 0);
                __builtin_prefetch(adjm + gp, 0, 0);
            }
        }
        __syncthreads();

        // B fragment: two 8-half runs (k = 8h.., 16+8h..) -> two b128 loads
        v16h bf;
        {
            const _Float16* bp = bT + (nb + mrow) * BT_STR + khalf * 8;
            h8 blo = *(const h8*)(bp);
            h8 bhi = *(const h8*)(bp + 16);
            bf = __builtin_shufflevector(blo, bhi, 0, 1, 2, 3, 4, 5, 6, 7,
                                         8, 9, 10, 11, 12, 13, 14, 15);
        }
        // 8 M-subtiles per wave; A fragment = two b128 loads each
#pragma unroll
        for (int j = 0; j < 8; ++j) {
            int ms = (w >> 2) + 2 * j;     // subtile 0..15
            const _Float16* ap = aT + (ms * 16 + mrow) * A_STR + khalf * 8;
            h8 alo = *(const h8*)(ap);
            h8 ahi = *(const h8*)(ap + 16);
            v16h af = __builtin_shufflevector(alo, ahi, 0, 1, 2, 3, 4, 5, 6, 7,
                                              8, 9, 10, 11, 12, 13, 14, 15);
            acc[j] = __builtin_amdgcn_wmma_f32_16x16x32_f16(
                false, af, false, bf, (short)0, acc[j], false, false);
        }
        __syncthreads();
    }

    // Scale by 1/relation_num, store f16 candidates
    const int col = n0 + nb + mrow;
    const float rs = rinv[col];
#pragma unroll
    for (int j = 0; j < 8; ++j) {
        int ms = (w >> 2) + 2 * j;
#pragma unroll
        for (int r = 0; r < 8; ++r) {
            int row = ms * 16 + r + 8 * khalf;
            candH[(size_t)row * LDIM + col] = (_Float16)(acc[j][r] * rs);
        }
    }
}

// ---------------------------------------------------------------------------
// Kernel 3: smoothed = similarities @ candH (WMMA, K=256), fused with
// diff^2 partial row sums. grid.x = L/16 = 256. No atomics: partial[row][blk].
// ---------------------------------------------------------------------------
__global__ __launch_bounds__(256) void k_loss_partial(const float* __restrict__ predicts,
                                                      const float* __restrict__ sims,
                                                      const _Float16* __restrict__ candH,
                                                      float* __restrict__ partial) {
    __shared__ __align__(16) _Float16 aT[256 * A_STR];  // similarities tile 256x32
    __shared__ __align__(16) _Float16 bT[16 * BT_STR];  // cand tile transposed [n][k]

    const int t     = threadIdx.x;
    const int lane  = t & 31;
    const int w     = t >> 5;
    const int n0    = blockIdx.x * 16;
    const int mrow  = lane & 15;
    const int khalf = lane >> 4;

    const v8f vzero = {0.f, 0.f, 0.f, 0.f, 0.f, 0.f, 0.f, 0.f};
    v8f acc0 = vzero, acc1 = vzero;

    for (int k0 = 0; k0 < BDIM; k0 += 32) {
#pragma unroll
        for (int i = 0; i < 16; ++i) {
            int e   = i * 256 + t;
            int row = e >> 4;
            int kd  = e & 15;
            float2 p = *(const float2*)(sims + (size_t)row * BDIM + k0 + 2 * kd);
            h2 hp;
            hp[0] = (_Float16)p.x;
            hp[1] = (_Float16)p.y;
            *(h2*)(aT + row * A_STR + 2 * kd) = hp;
        }
#pragma unroll
        for (int i = 0; i < 2; ++i) {
            int e = i * 256 + t;
            int k = e >> 4;
            int n = e & 15;
            bT[n * BT_STR + k] = candH[(size_t)(k0 + k) * LDIM + n0 + n];
        }
        __syncthreads();

        v16h bf;
        {
            const _Float16* bp = bT + mrow * BT_STR + khalf * 8;
            h8 blo = *(const h8*)(bp);
            h8 bhi = *(const h8*)(bp + 16);
            bf = __builtin_shufflevector(blo, bhi, 0, 1, 2, 3, 4, 5, 6, 7,
                                         8, 9, 10, 11, 12, 13, 14, 15);
        }
        {
            const _Float16* ap0 = aT + (w * 16 + mrow) * A_STR + khalf * 8;
            h8 a0lo = *(const h8*)(ap0);
            h8 a0hi = *(const h8*)(ap0 + 16);
            v16h af0 = __builtin_shufflevector(a0lo, a0hi, 0, 1, 2, 3, 4, 5, 6, 7,
                                               8, 9, 10, 11, 12, 13, 14, 15);
            const _Float16* ap1 = aT + ((w + 8) * 16 + mrow) * A_STR + khalf * 8;
            h8 a1lo = *(const h8*)(ap1);
            h8 a1hi = *(const h8*)(ap1 + 16);
            v16h af1 = __builtin_shufflevector(a1lo, a1hi, 0, 1, 2, 3, 4, 5, 6, 7,
                                               8, 9, 10, 11, 12, 13, 14, 15);
            acc0 = __builtin_amdgcn_wmma_f32_16x16x32_f16(false, af0, false, bf,
                                                          (short)0, acc0, false, false);
            acc1 = __builtin_amdgcn_wmma_f32_16x16x32_f16(false, af1, false, bf,
                                                          (short)0, acc1, false, false);
        }
        __syncthreads();
    }

    // diff^2, reduce over the 16 columns handled by this half-wave
    const int col = n0 + mrow;
#pragma unroll
    for (int half = 0; half < 2; ++half) {
        const v8f& a = half ? acc1 : acc0;
        const int mb = (half ? (w + 8) : w) * 16;
#pragma unroll
        for (int r = 0; r < 8; ++r) {
            int row = mb + r + 8 * khalf;
            float d  = predicts[(size_t)row * LDIM + col] - a[r];
            float sq = d * d;
            sq += __shfl_xor(sq, 1, 32);
            sq += __shfl_xor(sq, 2, 32);
            sq += __shfl_xor(sq, 4, 32);
            sq += __shfl_xor(sq, 8, 32);
            if (mrow == 0)
                partial[(size_t)row * 256 + blockIdx.x] = sq;
        }
    }
}

// ---------------------------------------------------------------------------
// Kernel 4: loss = sum(valid_i * sqrt(rowsq_i)) / sum(valid_i)
// ---------------------------------------------------------------------------
__global__ __launch_bounds__(256) void k_final(const float* __restrict__ sims,
                                               const float* __restrict__ partial,
                                               float* __restrict__ out) {
    __shared__ float sv[256];
    __shared__ float sc[256];
    const int i = threadIdx.x;

    float s = 0.0f;
    for (int j = 0; j < BDIM; ++j) s += sims[(size_t)i * BDIM + j];
    float rq = 0.0f;
    for (int b = 0; b < 256; ++b) rq += partial[(size_t)i * 256 + b];

    float valid = (s != 0.0f) ? 1.0f : 0.0f;
    sv[i] = valid * sqrtf(rq);
    sc[i] = valid;
    __syncthreads();
    for (int st = 128; st > 0; st >>= 1) {
        if (i < st) { sv[i] += sv[i + st]; sc[i] += sc[i + st]; }
        __syncthreads();
    }
    if (i == 0) out[0] = sv[0] / sc[0];
}

// ---------------------------------------------------------------------------
extern "C" void kernel_launch(void* const* d_in, const int* in_sizes, int n_in,
                              void* d_out, int out_size, void* d_ws, size_t ws_size,
                              hipStream_t stream) {
    const float*         predicts = (const float*)d_in[0];          // [B,L]
    const float*         sims     = (const float*)d_in[1];          // [B,B]
    const float*         adjv     = (const float*)d_in[2];          // [L,L]
    const unsigned char* adjm     = (const unsigned char*)d_in[3];  // [L,L] bool
    float* out = (float*)d_out;

    char* ws = (char*)d_ws;
    float*     rinv    = (float*)(ws);                 // L * 4        = 16 KB
    float*     partial = (float*)(ws + 16384);         // B * 256 * 4  = 256 KB
    _Float16*  candH   = (_Float16*)(ws + 524288);     // B * L * 2    = 2 MB

    k_relnum<<<LDIM / (256 * 4), 256, 0, stream>>>(adjm, rinv);
    k_cand<<<LDIM / 64, 256, 0, stream>>>(predicts, adjv, adjm, rinv, candH);
    k_loss_partial<<<LDIM / 16, 256, 0, stream>>>(predicts, sims, candH, partial);
    k_final<<<1, 256, 0, stream>>>(sims, partial, out);
}
